// MultiScaleAttention_68788196213132
// MI455X (gfx1250) — matrix operand order repaired and checked by reading, not compiled, not run on hardware
//
#include <hip/hip_runtime.h>
#include <hip/hip_bf16.h>

#define B_ 4
#define H_ 56
#define W_ 56
#define L_ (H_*W_)          // 3136
#define N_ (L_+1)           // 3137
#define NPAD 3168           // 99*32, padded token count
#define DIM_ 96
#define KCH 99              // key chunks of 32
#define QT 197              // ceil(N/16) query tiles
#define SCALE_ 0.10206207261596575f   // 96^-0.5
#define EPS_ 1e-5f

typedef __attribute__((ext_vector_type(16))) _Float16 v16h;
typedef __attribute__((ext_vector_type(8)))  _Float16 v8h;
typedef __attribute__((ext_vector_type(8)))  float    v8f;

static __device__ inline v8f wmma32(v16h a, v16h b, v8f c) {
  // D = A(16x32 f16) * B(32x16 f16) + C(16x16 f32)
  return __builtin_amdgcn_wmma_f32_16x16x32_f16(false, a, false, b, (short)0, c,
                                                false, false);
}

// A-fragment (16x32 f16): lane row = l&15; halves j: K = 2*kb+j (contiguous), kb = 8*(l>>4)
static __device__ inline v16h loadA_h(const _Float16* rowp, int kb) {
  v8h lo = *(const v8h*)(rowp + 2 * kb);
  v8h hi = *(const v8h*)(rowp + 2 * kb + 8);
  return __builtin_shufflevector(lo, hi, 0,1,2,3,4,5,6,7,8,9,10,11,12,13,14,15);
}
static __device__ inline v16h loadA_f(const float* rowp, int kb) {
  v16h a;
#pragma unroll
  for (int j = 0; j < 8; ++j)  a[j] = (_Float16)rowp[2 * kb + j];
#pragma unroll
  for (int j = 8; j < 16; ++j) a[j] = (_Float16)rowp[2 * kb + j];
  return a;
}
// B-fragment (32x16 f16): lane col = l&15; halves j: K = 16*(l>>4)+j (contiguous)
static __device__ inline v16h loadB_f(const float* colp, int kh) {
  v16h b;
#pragma unroll
  for (int j = 0; j < 16; ++j) b[j] = (_Float16)colp[kh + j];
  return b;
}

// ---------------- kernel 1: qkv = x @ qkv_w^T, scattered to q/k/v -------------
__global__ __launch_bounds__(128) void k_qkv(const float* __restrict__ x,
                                             const float* __restrict__ w,
                                             float* __restrict__ qr,
                                             float* __restrict__ kr,
                                             float* __restrict__ vr) {
  const int M  = B_ * N_;          // 12548
  const int MT = (M + 15) / 16;    // 785
  int tile = blockIdx.x * 4 + (threadIdx.x >> 5);
  if (tile >= MT * 18) return;
  int mt = tile / 18, nt = tile % 18;
  int lane = threadIdx.x & 31;
  int n16 = lane & 15, half = lane >> 4;
  int arow = mt * 16 + n16; if (arow >= M) arow = M - 1;
  const float* ap = x + (size_t)arow * DIM_;
  int wrow = nt * 16 + n16;        // 0..287 (output column)
  const float* bp = w + (size_t)wrow * DIM_;
  v8f acc = (v8f)0.0f;
#pragma unroll
  for (int kc = 0; kc < 96; kc += 32)
    acc = wmma32(loadA_f(ap + kc, half * 8), loadB_f(bp + kc, half * 16), acc);
  int tt = wrow / 96, cc = wrow % 96;
  float* dst = (tt == 0) ? qr : (tt == 1) ? kr : vr;
#pragma unroll
  for (int r = 0; r < 8; ++r) {
    int row = mt * 16 + r + 8 * half;
    if (row < M) dst[(size_t)row * DIM_ + cc] = acc[r];
  }
}

// ------------- kernel 2: depthwise 3x3 pool + LayerNorm (per token) ----------
__global__ __launch_bounds__(128) void k_poolnorm(const float* __restrict__ in,
                                                  const float* __restrict__ cw,
                                                  const float* __restrict__ gw,
                                                  const float* __restrict__ gb,
                                                  float* __restrict__ out32,
                                                  _Float16* __restrict__ rowh,
                                                  _Float16* __restrict__ colh) {
  int b = blockIdx.x / N_;
  int tok = blockIdx.x % N_;
  int c = threadIdx.x;
  float y = 0.f;
  if (c < DIM_) {
    if (tok == 0) {
      y = in[(size_t)(b * N_) * DIM_ + c];
    } else {
      int j = tok - 1, y0 = j / W_, x0 = j % W_;
#pragma unroll
      for (int dy = 0; dy < 3; ++dy) {
        int yy = y0 + dy - 1; if (yy < 0 || yy >= H_) continue;
#pragma unroll
        for (int dx = 0; dx < 3; ++dx) {
          int xx = x0 + dx - 1; if (xx < 0 || xx >= W_) continue;
          y += in[(size_t)(b * N_ + 1 + yy * W_ + xx) * DIM_ + c] *
               cw[c * 9 + dy * 3 + dx];
        }
      }
    }
  }
  __shared__ float sred[128];
  sred[threadIdx.x] = (c < DIM_) ? y : 0.f;
  __syncthreads();
  for (int s = 64; s > 0; s >>= 1) {
    if (threadIdx.x < s) sred[threadIdx.x] += sred[threadIdx.x + s];
    __syncthreads();
  }
  float mu = sred[0] * (1.f / DIM_);
  __syncthreads();
  float d = (c < DIM_) ? (y - mu) : 0.f;
  sred[threadIdx.x] = d * d;
  __syncthreads();
  for (int s = 64; s > 0; s >>= 1) {
    if (threadIdx.x < s) sred[threadIdx.x] += sred[threadIdx.x + s];
    __syncthreads();
  }
  float var = sred[0] * (1.f / DIM_);
  if (c < DIM_) {
    float o = (y - mu) * rsqrtf(var + EPS_) * gw[c] + gb[c];
    if (out32) out32[(size_t)(b * N_ + tok) * DIM_ + c] = o;
    if (rowh)  rowh[(size_t)(b * NPAD + tok) * DIM_ + c] = (_Float16)o;
    if (colh)  colh[(size_t)(b * DIM_ + c) * NPAD + tok] = (_Float16)o;
  }
}

// ----------- kernel 3: decomposed relative-position bias projections --------
__global__ __launch_bounds__(128) void k_rel(const float* __restrict__ qp,
                                             const float* __restrict__ rph,
                                             const float* __restrict__ rpw,
                                             float* __restrict__ relh,
                                             float* __restrict__ relw) {
  int b = blockIdx.x / L_;
  int j = blockIdx.x % L_;
  int y0 = j / W_, x0 = j % W_;
  __shared__ float qrow[DIM_];
  if (threadIdx.x < DIM_)
    qrow[threadIdx.x] = qp[(size_t)(b * N_ + 1 + j) * DIM_ + threadIdx.x];
  __syncthreads();
  int t = threadIdx.x;
  if (t < H_) {
    const float* rp = rph + (size_t)(y0 - t + (H_ - 1)) * DIM_;
    float s = 0.f;
    for (int c = 0; c < DIM_; ++c) s += qrow[c] * rp[c];
    relh[(size_t)(b * L_ + j) * H_ + t] = s;
  } else if (t >= 64 && t < 64 + W_) {
    int k = t - 64;
    const float* rp = rpw + (size_t)(x0 - k + (W_ - 1)) * DIM_;
    float s = 0.f;
    for (int c = 0; c < DIM_; ++c) s += qrow[c] * rp[c];
    relw[(size_t)(b * L_ + j) * W_ + k] = s;
  }
}

// --------- kernel 4: fused flash attention + bias + residual ----------------
__global__ __launch_bounds__(128) void k_attn(const _Float16* __restrict__ qh,
                                              const _Float16* __restrict__ kh,
                                              const _Float16* __restrict__ vt,
                                              const float* __restrict__ qp,
                                              const float* __restrict__ relh,
                                              const float* __restrict__ relw,
                                              float* __restrict__ ao) {
  __shared__ __align__(64) _Float16 Pb[4][16 * 32];
  int wid = threadIdx.x >> 5;
  int tile = blockIdx.x * 4 + wid;
  if (tile >= B_ * QT) return;
  int b = tile / QT, qt = tile % QT;
  int lane = threadIdx.x & 31;
  int n16 = lane & 15, half = lane >> 4;
  int q0 = qt * 16;

  // Q fragments (A-layout): lane row = q0 + n16
  const _Float16* qhp = qh + (size_t)(b * NPAD + q0 + n16) * DIM_;
  v16h aQ0 = loadA_h(qhp + 0,  half * 8);
  v16h aQ1 = loadA_h(qhp + 32, half * 8);
  v16h aQ2 = loadA_h(qhp + 64, half * 8);

  // per-C-row query metadata (rows m = r + 8*half)
  const float* rhrow[8];
  const float* rwrow[8];
  bool qv[8];
#pragma unroll
  for (int r = 0; r < 8; ++r) {
    int qi = q0 + r + 8 * half;
    bool v = (qi >= 1 && qi < N_);
    qv[r] = v;
    int qq = v ? (qi - 1) : 0;
    rhrow[r] = relh + (size_t)(b * L_ + qq) * H_;
    rwrow[r] = relw + (size_t)(b * L_ + qq) * W_;
  }

  float mrun[8], lrun[8];
  v8f O[6];
#pragma unroll
  for (int r = 0; r < 8; ++r) { mrun[r] = -__builtin_inff(); lrun[r] = 0.f; }
#pragma unroll
  for (int ct = 0; ct < 6; ++ct) O[ct] = (v8f)0.0f;
  _Float16* myP = &Pb[wid][0];

  for (int kc2 = 0; kc2 < KCH; ++kc2) {
    int k0 = kc2 * 32;
    float s0v[8], s1v[8], p0[8], p1[8];
#pragma unroll
    for (int t = 0; t < 2; ++t) {
      int kt = k0 + t * 16;
      const _Float16* khp = kh + (size_t)(b * NPAD + kt + n16) * DIM_ + half * 16;
      v16h bK0 = *(const v16h*)(khp + 0);
      v16h bK1 = *(const v16h*)(khp + 32);
      v16h bK2 = *(const v16h*)(khp + 64);
      v8f s = (v8f)0.0f;
      s = wmma32(aQ0, bK0, s);
      s = wmma32(aQ1, bK1, s);
      s = wmma32(aQ2, bK2, s);
      int ki = kt + n16;
      bool kval = ki < N_;
      bool kb = kval && (ki >= 1);
      int khr = 0, kwr = 0;
      if (kb) { int kk = ki - 1; khr = kk / W_; kwr = kk - khr * W_; }
      float* sv = t ? s1v : s0v;
#pragma unroll
      for (int r = 0; r < 8; ++r) {
        float v;
        if (!kval) v = -__builtin_inff();
        else {
          v = s[r] * SCALE_;
          if (kb && qv[r]) v += rhrow[r][khr] + rwrow[r][kwr];
        }
        sv[r] = v;
      }
    }
    // online softmax (rows live in 16-lane halves; xor masks 1..8 stay inside)
#pragma unroll
    for (int r = 0; r < 8; ++r) {
      float v = fmaxf(s0v[r], s1v[r]);
#pragma unroll
      for (int m = 1; m < 16; m <<= 1) v = fmaxf(v, __shfl_xor(v, m, 32));
      float mn = fmaxf(mrun[r], v);
      float al = __expf(mrun[r] - mn);
      mrun[r] = mn;
      p0[r] = __expf(s0v[r] - mn);
      p1[r] = __expf(s1v[r] - mn);
      float sum = p0[r] + p1[r];
#pragma unroll
      for (int m = 1; m < 16; m <<= 1) sum += __shfl_xor(sum, m, 32);
      lrun[r] = lrun[r] * al + sum;
#pragma unroll
      for (int ct = 0; ct < 6; ++ct) O[ct][r] *= al;
    }
    // transpose P (C-layout -> A-layout) through wave-private LDS
#pragma unroll
    for (int r = 0; r < 8; ++r) {
      int m = r + 8 * half;
      myP[m * 32 + n16]      = (_Float16)p0[r];
      myP[m * 32 + 16 + n16] = (_Float16)p1[r];
    }
    asm volatile("s_wait_dscnt 0" ::: "memory");
    v16h aP = loadA_h(myP + n16 * 32, half * 8);
    // P (16x32) * V (32x96): vt is key-major -> contiguous B fragments
#pragma unroll
    for (int ct = 0; ct < 6; ++ct) {
      const _Float16* vp =
          vt + (size_t)(b * DIM_ + ct * 16 + n16) * NPAD + k0 + half * 16;
      v16h bV = *(const v16h*)vp;
      O[ct] = wmma32(aP, bV, O[ct]);
    }
  }
  // epilogue: 1/l scale + q residual (tokens >= 1)
#pragma unroll
  for (int r = 0; r < 8; ++r) {
    int qi = q0 + r + 8 * half;
    if (qi >= N_) continue;
    float inv = 1.f / lrun[r];
    const float* qres = qp + (size_t)(b * N_ + qi) * DIM_;
#pragma unroll
    for (int ct = 0; ct < 6; ++ct) {
      int cch = ct * 16 + n16;
      float o = O[ct][r] * inv;
      if (qi >= 1) o += qres[cch];
      ao[(size_t)(b * N_ + qi) * DIM_ + cch] = o;
    }
  }
}

// ---------------- kernel 5: out = ao @ proj_w^T + proj_b --------------------
__global__ __launch_bounds__(128) void k_proj(const float* __restrict__ ain,
                                              const float* __restrict__ w,
                                              const float* __restrict__ bias,
                                              float* __restrict__ out) {
  const int M  = B_ * N_;
  const int MT = (M + 15) / 16;
  int tile = blockIdx.x * 4 + (threadIdx.x >> 5);
  if (tile >= MT * 6) return;
  int mt = tile / 6, nt = tile % 6;
  int lane = threadIdx.x & 31;
  int n16 = lane & 15, half = lane >> 4;
  int arow = mt * 16 + n16; if (arow >= M) arow = M - 1;
  const float* ap = ain + (size_t)arow * DIM_;
  int wrow = nt * 16 + n16;
  const float* bp = w + (size_t)wrow * DIM_;
  v8f acc = (v8f)0.0f;
#pragma unroll
  for (int kc = 0; kc < 96; kc += 32)
    acc = wmma32(loadA_f(ap + kc, half * 8), loadB_f(bp + kc, half * 16), acc);
  float bb = bias[wrow];
#pragma unroll
  for (int r = 0; r < 8; ++r) {
    int row = mt * 16 + r + 8 * half;
    if (row < M) out[(size_t)row * DIM_ + wrow] = acc[r] + bb;
  }
}

extern "C" void kernel_launch(void* const* d_in, const int* in_sizes, int n_in,
                              void* d_out, int out_size, void* d_ws, size_t ws_size,
                              hipStream_t stream) {
  const float* x      = (const float*)d_in[0];
  const float* qkv_w  = (const float*)d_in[1];
  const float* proj_w = (const float*)d_in[2];
  const float* proj_b = (const float*)d_in[3];
  const float* pool_q = (const float*)d_in[4];
  const float* nqw    = (const float*)d_in[5];
  const float* nqb    = (const float*)d_in[6];
  const float* pool_k = (const float*)d_in[7];
  const float* nkw    = (const float*)d_in[8];
  const float* nkb    = (const float*)d_in[9];
  const float* pool_v = (const float*)d_in[10];
  const float* nvw    = (const float*)d_in[11];
  const float* nvb    = (const float*)d_in[12];
  const float* rph    = (const float*)d_in[13];
  const float* rpw    = (const float*)d_in[14];

  char* ws = (char*)d_ws;
  size_t off = 0;
  auto alloc = [&](size_t bytes) -> char* {
    char* p = ws + off;
    off += (bytes + 255) & ~(size_t)255;
    return p;
  };
  const size_t BN96f  = (size_t)B_ * N_ * DIM_ * sizeof(float);
  const size_t BNP96h = (size_t)B_ * NPAD * DIM_ * sizeof(_Float16);
  float*    q_raw  = (float*)alloc(BN96f);
  float*    k_raw  = (float*)alloc(BN96f);
  float*    v_raw  = (float*)alloc(BN96f);
  float*    q_pool = (float*)alloc(BN96f);
  _Float16* qh     = (_Float16*)alloc(BNP96h);
  _Float16* kh     = (_Float16*)alloc(BNP96h);
  _Float16* vt     = (_Float16*)alloc(BNP96h);
  float*    relh   = (float*)alloc((size_t)B_ * L_ * H_ * sizeof(float));
  float*    relw   = (float*)alloc((size_t)B_ * L_ * W_ * sizeof(float));
  float*    ao     = (float*)alloc(BN96f);

  // zero the padded token rows/cols (capture-safe async memsets)
  hipMemsetAsync(qh, 0, BNP96h, stream);
  hipMemsetAsync(kh, 0, BNP96h, stream);
  hipMemsetAsync(vt, 0, BNP96h, stream);

  {
    int tiles = ((B_ * N_ + 15) / 16) * 18;
    k_qkv<<<(tiles + 3) / 4, 128, 0, stream>>>(x, qkv_w, q_raw, k_raw, v_raw);
  }
  k_poolnorm<<<B_ * N_, 128, 0, stream>>>(q_raw, pool_q, nqw, nqb, q_pool, qh, nullptr);
  k_poolnorm<<<B_ * N_, 128, 0, stream>>>(k_raw, pool_k, nkw, nkb, nullptr, kh, nullptr);
  k_poolnorm<<<B_ * N_, 128, 0, stream>>>(v_raw, pool_v, nvw, nvb, nullptr, nullptr, vt);
  k_rel<<<B_ * L_, 128, 0, stream>>>(q_pool, rph, rpw, relh, relw);
  k_attn<<<(B_ * QT + 3) / 4, 128, 0, stream>>>(qh, kh, vt, q_pool, relh, relw, ao);
  {
    int tiles = ((B_ * N_ + 15) / 16) * 6;
    k_proj<<<(tiles + 3) / 4, 128, 0, stream>>>(ao, proj_w, proj_b, (float*)d_out);
  }
}
